// GCNRegressionModel_36498632081445
// MI455X (gfx1250) — matrix operand order repaired and checked by reading, not compiled, not run on hardware
//
#include <hip/hip_runtime.h>
#include <hip/hip_bf16.h>

typedef __attribute__((ext_vector_type(16))) _Float16 v16h;
typedef __attribute__((ext_vector_type(8)))  float    v8f;

// ---------------------------------------------------------------------------
// Utility elementwise kernels
// ---------------------------------------------------------------------------
__global__ void zero_f32(float* __restrict__ p, long long n) {
    long long i = (long long)blockIdx.x * blockDim.x + threadIdx.x;
    if (i < n) p[i] = 0.0f;
}

__global__ void deg_accum(const int* __restrict__ dst, float* __restrict__ deg, int E) {
    int e = blockIdx.x * blockDim.x + threadIdx.x;
    if (e < E) atomicAdd(&deg[dst[e]], 1.0f);
}

__global__ void finish_dinv(float* __restrict__ deg, int n) {
    int i = blockIdx.x * blockDim.x + threadIdx.x;
    if (i < n) deg[i] = rsqrtf(deg[i] + 1.0f);   // deg buffer becomes dinv in place
}

// ---------------------------------------------------------------------------
// Edge scatter: agg[dst] += h[src] * dinv[src]*dinv[dst]; one thread = 4 feats.
// A wave's lanes cover contiguous feature chunks of one (or few) edges:
// index loads are broadcast, h gathers / atomics are coalesced 128B-512B.
// ---------------------------------------------------------------------------
__global__ void scatter_edges(const float* __restrict__ h, const float* __restrict__ dinv,
                              const int* __restrict__ src, const int* __restrict__ dst,
                              float* __restrict__ agg, int E, int F4, int F) {
    long long tid = (long long)blockIdx.x * blockDim.x + threadIdx.x;
    int e = (int)(tid / F4);
    if (e >= E) return;
    int c = (int)(tid - (long long)e * F4) * 4;
    int s = src[e], d = dst[e];
    float norm = dinv[s] * dinv[d];
    const float4 v = *reinterpret_cast<const float4*>(h + (long long)s * F + c);
    float* ap = agg + (long long)d * F + c;
    atomicAdd(ap + 0, v.x * norm);
    atomicAdd(ap + 1, v.y * norm);
    atomicAdd(ap + 2, v.z * norm);
    atomicAdd(ap + 3, v.w * norm);
}

// ---------------------------------------------------------------------------
// Fused WMMA GEMM + self-loop/bias epilogue, 2-wide N register tiling.
//   H   = relu?(X) @ W                      (stored for the edge gather)
//   AGG = H * dinv^2 + bias                 (self-loop init of the scatter acc)
// W staged into LDS pre-swizzled into B-fragment order (one 32B ds load per
// fragment). Each wave computes a 16x32 output strip: one A fragment feeds
// two back-to-back WMMAs, halving A traffic and f32->f16 conversion work.
// ---------------------------------------------------------------------------
template <int K, bool RELU>
__global__ __launch_bounds__(256)
void gemm_fused(const float* __restrict__ X, const float* __restrict__ W,
                const float* __restrict__ dinv, const float* __restrict__ bias,
                float* __restrict__ H, float* __restrict__ AGG, int M, int F) {
    constexpr int KB = K / 32;                    // k-blocks
    extern __shared__ __align__(32) _Float16 wl[];

    // ---- stage W (f32 -> f16) into LDS in B-fragment order ----
    // wl[((tileN*KB + kb)*32 + lane)*16 + j] =
    //     W[(kb*32 + (lane>>4)*16 + j)*F + tileN*16 + (lane&15)]
    const int total = K * F;                      // always a multiple of 512 here
    for (int i = threadIdx.x; i < total; i += 256) {
        int j     = i & 15;
        int lane2 = (i >> 4) & 31;
        int fidx  = i >> 9;
        int kb    = fidx % KB;
        int tN    = fidx / KB;
        int sk    = kb * 32 + (lane2 >> 4) * 16 + j;
        int sf    = tN * 16 + (lane2 & 15);
        wl[i] = (_Float16)W[(long long)sk * F + sf];
    }
    __syncthreads();

    const int lane   = threadIdx.x & 31;
    const int wave   = blockIdx.x * (blockDim.x >> 5) + (threadIdx.x >> 5);
    const int npairs = F >> 5;                    // N-tile pairs (F/32)
    const int mtiles = (M + 15) >> 4;
    const int tileN0 = (wave % npairs) * 2;       // first of two adjacent N tiles
    const int tileM  = wave / npairs;
    if (tileM >= mtiles) return;                  // wave-uniform exit (after sync)

    const int lo   = lane & 15;
    const int hi   = lane >> 4;
    const int row  = tileM * 16 + lo;
    const int col0 = tileN0 * 16 + lo;
    const bool rowOK = (row < M);

    v8f acc0 = {}, acc1 = {};
#pragma unroll
    for (int kk = 0; kk < K; kk += 32) {
        // ---- A fragment: 16x32 f16, contiguous k = 16*hi + j, as in B; optional fused ReLU on load ----
        v16h a;
        if (rowOK) {
            const float* xr = X + (long long)row * K + kk + hi * 16;
#pragma unroll
            for (int j = 0; j < 8; ++j) {
                float v0 = xr[j], v1 = xr[8 + j];
                if (RELU) { v0 = fmaxf(v0, 0.0f); v1 = fmaxf(v1, 0.0f); }
                a[j] = (_Float16)v0; a[8 + j] = (_Float16)v1;
            }
        } else {
#pragma unroll
            for (int j = 0; j < 16; ++j) a[j] = (_Float16)0.0f;
        }
        // ---- two B fragments: contiguous 32B LDS loads ----
        const int kb = kk >> 5;
        const v16h bf0 = *reinterpret_cast<const v16h*>(
            &wl[((((tileN0 + 0) * KB + kb) * 32 + lane) << 4)]);
        const v16h bf1 = *reinterpret_cast<const v16h*>(
            &wl[((((tileN0 + 1) * KB + kb) * 32 + lane) << 4)]);

        acc0 = __builtin_amdgcn_wmma_f32_16x16x32_f16(
            false, a, false, bf0, (short)0, acc0, false, false);
        acc1 = __builtin_amdgcn_wmma_f32_16x16x32_f16(
            false, a, false, bf1, (short)0, acc1, false, false);
    }

    // ---- epilogue: store H and AGG = H*dinv^2 + bias for both columns ----
    const float bcol0 = bias[col0];
    const float bcol1 = bias[col0 + 16];
#pragma unroll
    for (int r = 0; r < 8; ++r) {
        int orow = tileM * 16 + r + hi * 8;
        if (orow < M) {
            float di2 = dinv[orow];
            di2 *= di2;
            long long idx = (long long)orow * F + col0;
            float v0 = acc0[r], v1 = acc1[r];
            H[idx]        = v0;
            H[idx + 16]   = v1;
            AGG[idx]      = v0 * di2 + bcol0;
            AGG[idx + 16] = v1 * di2 + bcol1;
        }
    }
}

// ---------------------------------------------------------------------------
// Global mean pool (with deferred ReLU of last layer) + FC
// ---------------------------------------------------------------------------
__global__ void pool_accum(const float* __restrict__ h, const int* __restrict__ batch,
                           float* __restrict__ pooled, float* __restrict__ cnt, int n) {
    long long tid = (long long)blockIdx.x * blockDim.x + threadIdx.x;
    int node = (int)(tid >> 5);
    int f    = (int)(tid & 31);
    if (node >= n) return;
    int g = batch[node];
    atomicAdd(&pooled[(long long)g * 32 + f], fmaxf(h[(long long)node * 32 + f], 0.0f));
    if (f == 0) atomicAdd(&cnt[g], 1.0f);
}

__global__ void final_fc(const float* __restrict__ pooled, const float* __restrict__ cnt,
                         const float* __restrict__ Wfc, const float* __restrict__ bfc,
                         float* __restrict__ out, int G) {
    int g = blockIdx.x * blockDim.x + threadIdx.x;
    if (g >= G) return;
    float inv = 1.0f / fmaxf(cnt[g], 1.0f);
    float acc = bfc[0];
#pragma unroll
    for (int f = 0; f < 32; ++f) acc += pooled[(long long)g * 32 + f] * inv * Wfc[f];
    out[g] = acc;
}

// ---------------------------------------------------------------------------
// Host driver
// ---------------------------------------------------------------------------
static inline unsigned blocks_for(long long n, int bs) {
    return (unsigned)((n + bs - 1) / bs);
}

extern "C" void kernel_launch(void* const* d_in, const int* in_sizes, int n_in,
                              void* d_out, int out_size, void* d_ws, size_t ws_size,
                              hipStream_t stream) {
    const float* x     = (const float*)d_in[0];
    const int*   ei    = (const int*)d_in[1];
    const int*   batch = (const int*)d_in[2];
    const float* W[4]  = { (const float*)d_in[4], (const float*)d_in[6],
                           (const float*)d_in[8], (const float*)d_in[10] };
    const float* b[4]  = { (const float*)d_in[5], (const float*)d_in[7],
                           (const float*)d_in[9], (const float*)d_in[11] };
    const float* Wfc = (const float*)d_in[12];
    const float* bfc = (const float*)d_in[13];
    float* out = (float*)d_out;

    const int n = in_sizes[0] / 64;     // 100000 nodes, IN_DIM=64
    const int E = in_sizes[1] / 2;      // 1.6M edges
    const int G = out_size;             // 256 graphs

    const int* src = ei;
    const int* dst = ei + E;

    // workspace layout (floats): 3 rotating feature buffers + dinv + pool
    float* b0     = (float*)d_ws;
    float* b1     = b0 + (long long)n * 128;
    float* b2     = b1 + (long long)n * 128;
    float* dinv   = b2 + (long long)n * 128;
    float* pooled = dinv + n;
    float* cnt    = pooled + (long long)G * 32;

    // ---- degrees -> dinv (recomputed every call; atomic accum needs clean start)
    zero_f32<<<blocks_for(n, 256), 256, 0, stream>>>(dinv, n);
    deg_accum<<<blocks_for(E, 256), 256, 0, stream>>>(dst, dinv, E);
    finish_dinv<<<blocks_for(n, 256), 256, 0, stream>>>(dinv, n);

    // buffer rotation: gemm writes H and AGG; AGG must not alias the gemm input
    const int   Ks[4]   = { 64, 128, 64, 64 };
    const int   Fs[4]   = { 128, 64, 64, 32 };
    const float* ins[4] = { x,  b1, b0, b2 };
    float*       hs[4]  = { b0, b2, b1, b0 };
    float*       ags[4] = { b1, b0, b2, b1 };

    for (int l = 0; l < 4; ++l) {
        const int K = Ks[l], F = Fs[l];
        const long long tiles = (long long)((n + 15) / 16) * (F / 32); // 16x32 strips
        const unsigned gblocks = blocks_for(tiles, 8);   // 8 waves (256 thr) / block
        const size_t lds = (size_t)K * F * sizeof(_Float16);
        if (l == 0)
            gemm_fused<64, false><<<gblocks, 256, lds, stream>>>(
                ins[l], W[l], dinv, b[l], hs[l], ags[l], n, F);
        else if (K == 128)
            gemm_fused<128, true><<<gblocks, 256, lds, stream>>>(
                ins[l], W[l], dinv, b[l], hs[l], ags[l], n, F);
        else
            gemm_fused<64, true><<<gblocks, 256, lds, stream>>>(
                ins[l], W[l], dinv, b[l], hs[l], ags[l], n, F);

        const int F4 = F / 4;
        scatter_edges<<<blocks_for((long long)E * F4, 256), 256, 0, stream>>>(
            hs[l], dinv, src, dst, ags[l], E, F4, F);
    }

    // ---- global mean pool (applies deferred ReLU) + FC ----
    zero_f32<<<blocks_for((long long)G * 32 + G, 256), 256, 0, stream>>>(
        pooled, (long long)G * 32 + G);   // pooled and cnt are contiguous
    pool_accum<<<blocks_for((long long)n * 32, 256), 256, 0, stream>>>(
        ags[3], batch, pooled, cnt, n);
    final_fc<<<blocks_for(G, 256), 256, 0, stream>>>(pooled, cnt, Wfc, bfc, out, G);
}
